// LiteEfficientAttention_74406013436281
// MI455X (gfx1250) — matrix operand order repaired and checked by reading, not compiled, not run on hardware
//
#include <hip/hip_runtime.h>
#include <hip/hip_fp16.h>
#include <cstdint>

typedef __attribute__((ext_vector_type(16))) _Float16 v16h;
typedef __attribute__((ext_vector_type(8)))  _Float16 v8h;
typedef __attribute__((ext_vector_type(8)))  float    v8f;

#define DEV __device__ __forceinline__

// ---- WMMA fragment helpers (wave32, 16x16x32 f16 -> f32) -------------------
// A fragment: 16x32 (MxK), A row-major with leading dim lda (elements).
// lane r=lane&15 holds row r; half h=lane>>4 selects K = 16h + 0..15 (contiguous).
DEV v16h load_a_rm(const _Float16* A, int lda, int lane) {
  int r = lane & 15, h = lane >> 4;
  const _Float16* p = A + (size_t)r * lda + h * 16;
  v8h a0 = *(const v8h*)p;
  v8h a1 = *(const v8h*)(p + 8);
  return __builtin_shufflevector(a0, a1, 0,1,2,3,4,5,6,7,8,9,10,11,12,13,14,15);
}
// B fragment: 32x16 (KxN), element (k,n) at Bp[n*ldb + k] (K contiguous per column).
// lane n=lane&15 holds column n; K = 16*(lane>>4) + 0..15 (contiguous).
DEV v16h load_b_cm(const _Float16* Bp, int ldb, int lane) {
  int n = lane & 15, h = lane >> 4;
  return *(const v16h*)(Bp + (size_t)n * ldb + h * 16);
}
DEV v8f wmma16(v16h a, v16h b, v8f c) {
  return __builtin_amdgcn_wmma_f32_16x16x32_f16(false, a, false, b, (short)0, c, false, false);
}
// Async global -> LDS 16-byte copy (CDNA5 ASYNCcnt path).
// lds generic-pointer low 32 bits == LDS byte offset (shared aperture base has
// zero low half), which is what the VDST operand of the async op wants.
DEV void stage_async16(const _Float16* gsrc, _Float16* lds_dst) {
  uint32_t l = (uint32_t)(uintptr_t)lds_dst;
  asm volatile("global_load_async_to_lds_b128 %0, %1, off"
               :: "v"(l), "v"(gsrc) : "memory");
}
DEV void wait_async0() {
  asm volatile("s_wait_asynccnt 0" ::: "memory");
}

// ---- problem constants -----------------------------------------------------
// B=8, C=384, H=W=48 -> N=2304, QK=192, SE=96
#define NB   8
#define NC   384
#define NN   2304
#define ND   192
#define NSE  96

// ---- K0: convert weights to fp16 (wq;wk;wv stacked -> wqkvh[768][384]) -----
__global__ void k_conv_weights(const float* __restrict__ wq, const float* __restrict__ wk,
                               const float* __restrict__ wv, const float* __restrict__ wproj,
                               _Float16* __restrict__ wqkvh, _Float16* __restrict__ wprojh) {
  int i = blockIdx.x * 256 + threadIdx.x;
  const int nq = ND * NC;       // 73728
  const int nv = NC * NC;       // 147456
  if (i < nq)                    wqkvh[i]              = (_Float16)wq[i];
  else if (i < 2 * nq)           wqkvh[i]              = (_Float16)wk[i - nq];
  else if (i < 2 * nq + nv)      wqkvh[i]              = (_Float16)wv[i - 2 * nq];
  else if (i < 2 * nq + 2 * nv)  wprojh[i - 2*nq - nv] = (_Float16)wproj[i - 2*nq - nv];
}

// ---- K1: x [b][c][n] f32 -> xh [b][n][c] f16 (tiled transpose) -------------
__global__ void k_x_transpose(const float* __restrict__ x, _Float16* __restrict__ xh) {
  __shared__ float tile[32][33];
  int b = blockIdx.z, n0 = blockIdx.x * 32, c0 = blockIdx.y * 32;
  int tx = threadIdx.x, ty = threadIdx.y;
  #pragma unroll
  for (int j = 0; j < 4; j++) {
    int c = c0 + ty + 8 * j;
    tile[ty + 8 * j][tx] = x[((size_t)(b * NC + c)) * NN + n0 + tx];
  }
  __syncthreads();
  #pragma unroll
  for (int j = 0; j < 4; j++) {
    int n = n0 + ty + 8 * j;
    xh[((size_t)(b * NN + n)) * NC + c0 + tx] = (_Float16)tile[tx][ty + 8 * j];
  }
}

// ---- K2: fused QKV projection GEMM -----------------------------------------
// QKV[768 x 2304] = Wqkv[768 x 384] @ X[384 x 2304] (+bias), per batch.
// grid (72 colblk, 6 rowblk, 8 b), block 256 (8 waves), wave = 16 rows x 32 cols.
__global__ void __launch_bounds__(256) k_qkv(const _Float16* __restrict__ wqkvh,
                                             const _Float16* __restrict__ xh,
                                             const float* __restrict__ bq, const float* __restrict__ bk,
                                             const float* __restrict__ bv,
                                             _Float16* __restrict__ qh, _Float16* __restrict__ kh,
                                             _Float16* __restrict__ vh) {
  int b = blockIdx.z;
  int n0 = blockIdx.x * 32;
  int r0 = blockIdx.y * 128 + (threadIdx.x >> 5) * 16;
  int lane = threadIdx.x & 31;
  const _Float16* Bbase = xh + ((size_t)b * NN + n0) * NC;
  v8f acc0 = {}; v8f acc1 = {};
  #pragma unroll 4
  for (int k = 0; k < NC; k += 32) {
    v16h a  = load_a_rm(wqkvh + (size_t)r0 * NC + k, NC, lane);
    v16h b0 = load_b_cm(Bbase + k, NC, lane);
    v16h b1 = load_b_cm(Bbase + (size_t)16 * NC + k, NC, lane);
    acc0 = wmma16(a, b0, acc0);
    acc1 = wmma16(a, b1, acc1);
  }
  int nl = lane & 15, ob = 8 * (lane >> 4);
  if (r0 < ND) {                         // Q -> qh[b][n][d], d contiguous
    int o0 = r0 + ob;
    float bs[8];
    #pragma unroll
    for (int i = 0; i < 8; i++) bs[i] = bq[o0 + i];
    v8h h0, h1;
    #pragma unroll
    for (int i = 0; i < 8; i++) { h0[i] = (_Float16)(acc0[i] + bs[i]); h1[i] = (_Float16)(acc1[i] + bs[i]); }
    *(v8h*)(qh + ((size_t)b * NN + n0 + nl)      * ND + o0) = h0;
    *(v8h*)(qh + ((size_t)b * NN + n0 + 16 + nl) * ND + o0) = h1;
  } else if (r0 < 2 * ND) {              // K -> kh[b][n][d]
    int o0 = r0 - ND + ob;
    float bs[8];
    #pragma unroll
    for (int i = 0; i < 8; i++) bs[i] = bk[o0 + i];
    v8h h0, h1;
    #pragma unroll
    for (int i = 0; i < 8; i++) { h0[i] = (_Float16)(acc0[i] + bs[i]); h1[i] = (_Float16)(acc1[i] + bs[i]); }
    *(v8h*)(kh + ((size_t)b * NN + n0 + nl)      * ND + o0) = h0;
    *(v8h*)(kh + ((size_t)b * NN + n0 + 16 + nl) * ND + o0) = h1;
  } else {                               // V -> vh[b][c][n], n contiguous
    int c0 = r0 - 2 * ND + ob;
    #pragma unroll
    for (int i = 0; i < 8; i++) {
      float bb = bv[c0 + i];
      vh[((size_t)(b * NC + c0 + i)) * NN + n0 + nl]      = (_Float16)(acc0[i] + bb);
      vh[((size_t)(b * NC + c0 + i)) * NN + n0 + 16 + nl] = (_Float16)(acc1[i] + bb);
    }
  }
}

// ---- K3: fused attention scores + softmax ----------------------------------
// One block (512 thr = 16 waves) owns a full 16-row stripe of scores:
// S[16 x 2304] lives in accumulators (each wave: 9 col tiles). Softmax is done
// with wave32 shfl_xor reductions within 16-lane halves + tiny LDS cross-wave.
__global__ void __launch_bounds__(512) k_attn(const _Float16* __restrict__ qh,
                                              const _Float16* __restrict__ kh,
                                              _Float16* __restrict__ p) {
  const float scale = 0.17677669529663687f;    // 32^-0.5
  __shared__ float red[16][16];                // [wave][row]
  int b = blockIdx.y, m0 = blockIdx.x * 16;
  int w = threadIdx.x >> 5, lane = threadIdx.x & 31;
  int ct0 = w * 9;                             // 16 waves * 9 tiles = 144 col tiles
  const _Float16* Qb = qh + ((size_t)b * NN + m0) * ND;
  const _Float16* Kb = kh + ((size_t)b * NN) * ND;
  v8f acc[9];
  #pragma unroll
  for (int t = 0; t < 9; t++) acc[t] = (v8f){};
  #pragma unroll
  for (int k = 0; k < ND; k += 32) {
    v16h a = load_a_rm(Qb + k, ND, lane);
    #pragma unroll
    for (int t = 0; t < 9; t++) {
      v16h bf = load_b_cm(Kb + (size_t)(ct0 + t) * 16 * ND + k, ND, lane);
      acc[t] = wmma16(a, bf, acc[t]);
    }
  }
  int half = lane >> 4, nl = lane & 15;
  #pragma unroll
  for (int t = 0; t < 9; t++)
    #pragma unroll
    for (int i = 0; i < 8; i++) acc[t][i] *= scale;
  // per-wave row max, reduce across 16-lane half with shfl_xor
  float pmax[8];
  #pragma unroll
  for (int i = 0; i < 8; i++) {
    float m = -3.402823466e38f;
    #pragma unroll
    for (int t = 0; t < 9; t++) m = fmaxf(m, acc[t][i]);
    #pragma unroll
    for (int mk = 8; mk >= 1; mk >>= 1) m = fmaxf(m, __shfl_xor(m, mk));
    pmax[i] = m;
  }
  if (nl == 0) {
    #pragma unroll
    for (int i = 0; i < 8; i++) red[w][8 * half + i] = pmax[i];
  }
  __syncthreads();
  float fmx[8];
  #pragma unroll
  for (int i = 0; i < 8; i++) {
    float m = -3.402823466e38f;
    for (int ww = 0; ww < 16; ww++) m = fmaxf(m, red[ww][8 * half + i]);
    fmx[i] = m;
  }
  __syncthreads();
  // exp + row sums
  float psum[8];
  #pragma unroll
  for (int i = 0; i < 8; i++) {
    float s = 0.0f;
    #pragma unroll
    for (int t = 0; t < 9; t++) {
      float e = __expf(acc[t][i] - fmx[i]);
      acc[t][i] = e;
      s += e;
    }
    #pragma unroll
    for (int mk = 8; mk >= 1; mk >>= 1) s += __shfl_xor(s, mk);
    psum[i] = s;
  }
  if (nl == 0) {
    #pragma unroll
    for (int i = 0; i < 8; i++) red[w][8 * half + i] = psum[i];
  }
  __syncthreads();
  float inv[8];
  #pragma unroll
  for (int i = 0; i < 8; i++) {
    float s = 0.0f;
    for (int ww = 0; ww < 16; ww++) s += red[ww][8 * half + i];
    inv[i] = 1.0f / s;
  }
  // normalized probabilities -> f16, row-major P[b][m][n]
  _Float16* Pb = p + ((size_t)b * NN + m0) * NN;
  #pragma unroll
  for (int t = 0; t < 9; t++)
    #pragma unroll
    for (int i = 0; i < 8; i++)
      Pb[(size_t)(8 * half + i) * NN + (ct0 + t) * 16 + nl] = (_Float16)(acc[t][i] * inv[i]);
}

// ---- K4: y = V @ P^T  (384 x 2304 per batch, K = 2304) ---------------------
// Block = 256 thr (8 waves) computes 128(c) x 32(m). The shared B operand
// (32 rows of P, the k-chunk) is staged once per block into LDS with
// global_load_async_to_lds_b128 (ASYNCcnt), double buffered so the async DMA
// overlaps WMMA on the previous chunk. B fragments then come from ds_load.
__global__ void __launch_bounds__(256) k_yv(const _Float16* __restrict__ vh,
                                            const _Float16* __restrict__ p,
                                            float* __restrict__ y) {
  __shared__ _Float16 sbuf[2][32 * 64];        // [buf][m_local(32)][k_local(64)] = 4KB each
  int b = blockIdx.z;
  int m0 = blockIdx.x * 32;
  int c0 = blockIdx.y * 128 + (threadIdx.x >> 5) * 16;
  int lane = threadIdx.x & 31;
  int tid = threadIdx.x;
  int r = tid >> 3, seg = tid & 7;             // 32 rows x 8 segs x 16B = 4KB per chunk
  const _Float16* A    = vh + ((size_t)(b * NC + c0)) * NN;
  const _Float16* Psrc = p + ((size_t)b * NN + m0 + r) * NN + seg * 8;
  int nl = lane & 15, h = lane >> 4;

  stage_async16(Psrc, &sbuf[0][r * 64 + seg * 8]);   // prologue: chunk 0
  v8f acc0 = {}; v8f acc1 = {};
  for (int c = 0; c < 36; ++c) {               // 36 chunks of K=64
    wait_async0();                             // own async loads for chunk c done
    __syncthreads();                           // everyone's done; prev reads retired
    if (c + 1 < 36)
      stage_async16(Psrc + (c + 1) * 64, &sbuf[(c + 1) & 1][r * 64 + seg * 8]);
    const _Float16* Bl = &sbuf[c & 1][0];
    int kg = c * 64;
    #pragma unroll
    for (int ks = 0; ks < 2; ++ks) {
      v16h a  = load_a_rm(A + kg + ks * 32, NN, lane);
      const _Float16* B0 = Bl + (size_t)nl * 64 + ks * 32 + h * 16;
      v16h b0 = *(const v16h*)B0;              // m tile 0 (cols m0+0..15)
      v16h b1 = *(const v16h*)(B0 + 16 * 64);  // m tile 1 (cols m0+16..31)
      acc0 = wmma16(a, b0, acc0);
      acc1 = wmma16(a, b1, acc1);
    }
  }
  int ob = 8 * h;
  #pragma unroll
  for (int i = 0; i < 8; i++) {
    y[((size_t)(b * NC + c0 + ob + i)) * NN + m0 + nl]      = acc0[i];
    y[((size_t)(b * NC + c0 + ob + i)) * NN + m0 + 16 + nl] = acc1[i];
  }
}

// ---- K5: SE squeeze-excite gate (tiny; scalar f32) -------------------------
__global__ void k_se(const float* __restrict__ y,
                     const float* __restrict__ wse1, const float* __restrict__ bse1,
                     const float* __restrict__ wse2, const float* __restrict__ bse2,
                     float* __restrict__ gate) {
  __shared__ float pooled[NC];
  __shared__ float hbuf[NSE];
  int b = blockIdx.x, t = threadIdx.x;   // 384 threads
  float s = 0.0f;
  const float* row = y + ((size_t)(b * NC + t)) * NN;
  for (int m = 0; m < NN; m++) s += row[m];
  pooled[t] = s * (1.0f / (float)NN);
  __syncthreads();
  if (t < NSE) {
    float a = bse1[t];
    const float* wr = wse1 + (size_t)t * NC;
    for (int c = 0; c < NC; c++) a += wr[c] * pooled[c];
    hbuf[t] = fmaxf(a, 0.0f);
  }
  __syncthreads();
  float g = bse2[t];
  const float* wr2 = wse2 + (size_t)t * NSE;
  for (int j = 0; j < NSE; j++) g += wr2[j] * hbuf[j];
  gate[b * NC + t] = 1.0f / (1.0f + __expf(-g));
}

// ---- K6: gated transpose: y[b][c][m]*gate[b][c] -> ygh[b][m][c] f16 --------
__global__ void k_gated_transpose(const float* __restrict__ y, const float* __restrict__ gate,
                                  _Float16* __restrict__ ygh) {
  __shared__ float tile[32][33];
  int b = blockIdx.z, m0 = blockIdx.x * 32, c0 = blockIdx.y * 32;
  int tx = threadIdx.x, ty = threadIdx.y;
  #pragma unroll
  for (int j = 0; j < 4; j++) {
    int c = c0 + ty + 8 * j;
    tile[ty + 8 * j][tx] = y[((size_t)(b * NC + c)) * NN + m0 + tx] * gate[b * NC + c];
  }
  __syncthreads();
  #pragma unroll
  for (int j = 0; j < 4; j++) {
    int m = m0 + ty + 8 * j;
    ygh[((size_t)(b * NN + m)) * NC + c0 + tx] = (_Float16)tile[tx][ty + 8 * j];
  }
}

// ---- K7: out = BN(Wproj @ (y*gate) + b_proj) -------------------------------
__global__ void __launch_bounds__(256) k_proj(const _Float16* __restrict__ wprojh,
                                              const _Float16* __restrict__ ygh,
                                              const float* __restrict__ bproj,
                                              const float* __restrict__ gamma, const float* __restrict__ beta,
                                              const float* __restrict__ mean, const float* __restrict__ var,
                                              float* __restrict__ out) {
  int b = blockIdx.z;
  int m0 = blockIdx.x * 16;
  int o0 = blockIdx.y * 128 + (threadIdx.x >> 5) * 16;
  int lane = threadIdx.x & 31;
  const _Float16* A  = wprojh + (size_t)o0 * NC;
  const _Float16* Bb = ygh + ((size_t)b * NN + m0) * NC;
  v8f acc = {};
  #pragma unroll 4
  for (int k = 0; k < NC; k += 32) {
    acc = wmma16(load_a_rm(A + k, NC, lane), load_b_cm(Bb + k, NC, lane), acc);
  }
  int nl = lane & 15, ob = 8 * (lane >> 4);
  #pragma unroll
  for (int i = 0; i < 8; i++) {
    int o = o0 + ob + i;
    float invs = gamma[o] * rsqrtf(var[o] + 1e-5f);
    float val  = acc[i] + bproj[o];
    out[((size_t)(b * NC + o)) * NN + m0 + nl] = (val - mean[o]) * invs + beta[o];
  }
}

// ---- host launcher ---------------------------------------------------------
extern "C" void kernel_launch(void* const* d_in, const int* in_sizes, int n_in,
                              void* d_out, int out_size, void* d_ws, size_t ws_size,
                              hipStream_t stream) {
  (void)in_sizes; (void)n_in; (void)out_size; (void)ws_size;
  const float* x     = (const float*)d_in[0];
  const float* wq    = (const float*)d_in[1];
  const float* bq    = (const float*)d_in[2];
  const float* wk    = (const float*)d_in[3];
  const float* bk    = (const float*)d_in[4];
  const float* wv    = (const float*)d_in[5];
  const float* bv    = (const float*)d_in[6];
  const float* wproj = (const float*)d_in[7];
  const float* bproj = (const float*)d_in[8];
  const float* gamma = (const float*)d_in[9];
  const float* beta  = (const float*)d_in[10];
  const float* mean  = (const float*)d_in[11];
  const float* var   = (const float*)d_in[12];
  const float* wse1  = (const float*)d_in[13];
  const float* bse1  = (const float*)d_in[14];
  const float* wse2  = (const float*)d_in[15];
  const float* bse2  = (const float*)d_in[16];
  float* out = (float*)d_out;

  char* ws = (char*)d_ws;
  size_t off = 0;
  auto wsalloc = [&](size_t bytes) -> void* {
    void* ptr = ws + off;
    off += (bytes + 255) & ~(size_t)255;
    return ptr;
  };
  _Float16* wqkvh  = (_Float16*)wsalloc((size_t)768 * NC * 2);
  _Float16* wprojh = (_Float16*)wsalloc((size_t)NC * NC * 2);
  _Float16* xh     = (_Float16*)wsalloc((size_t)NB * NN * NC * 2);
  _Float16* qh     = (_Float16*)wsalloc((size_t)NB * NN * ND * 2);
  _Float16* kh     = (_Float16*)wsalloc((size_t)NB * NN * ND * 2);
  _Float16* vh     = (_Float16*)wsalloc((size_t)NB * NC * NN * 2);
  _Float16* pp     = (_Float16*)wsalloc((size_t)NB * NN * NN * 2);
  float*    yb     = (float*)   wsalloc((size_t)NB * NC * NN * 4);
  float*    gate   = (float*)   wsalloc((size_t)NB * NC * 4);
  _Float16* ygh    = (_Float16*)wsalloc((size_t)NB * NN * NC * 2);

  k_conv_weights<<<1728, 256, 0, stream>>>(wq, wk, wv, wproj, wqkvh, wprojh);
  k_x_transpose<<<dim3(72, 12, NB), dim3(32, 8), 0, stream>>>(x, xh);
  k_qkv<<<dim3(72, 6, NB), 256, 0, stream>>>(wqkvh, xh, bq, bk, bv, qh, kh, vh);
  k_attn<<<dim3(144, NB), 512, 0, stream>>>(qh, kh, pp);
  k_yv<<<dim3(72, 3, NB), 256, 0, stream>>>(vh, pp, yb);
  k_se<<<NB, NC, 0, stream>>>(yb, wse1, bse1, wse2, bse2, gate);
  k_gated_transpose<<<dim3(72, 12, NB), dim3(32, 8), 0, stream>>>(yb, gate, ygh);
  k_proj<<<dim3(144, 3, NB), 256, 0, stream>>>(wprojh, ygh, bproj, gamma, beta, mean, var, out);
}
